// ConditionedSKOLRTransition_30305289240702
// MI455X (gfx1250) — compile-verified, matrix operand order checked by reading, not run on hardware
//
#include <hip/hip_runtime.h>
#include <hip/hip_bf16.h>

typedef __bf16 bf16_t;
typedef __attribute__((ext_vector_type(16))) __bf16 v16bf;
typedef __attribute__((ext_vector_type(8)))  __bf16 bf16x8;
typedef __attribute__((ext_vector_type(8)))  float  v8f;

#define N_BR 4
#define Dd   256
#define SDp  64
#define RH   64
#define MH   128
#define UD   16
#define NOBS 25
#define BSZ  32768
#define MEAS_IN 320   // Dd + SDp
// LDS staging strides (bf16 elems); all multiples of 8 (16B aligned rows),
// row stride mod 64 banks lands on distinct 4-bank groups per row.
#define H_STRIDE 136
#define G_STRIDE 72
#define Z_STRIDE 264
#define WAVE_LDS 4224  // 16 * Z_STRIDE (largest stage), reused for h/g/z

__device__ __forceinline__ v8f zero8() {
  v8f z;
#pragma unroll
  for (int i = 0; i < 8; ++i) z[i] = 0.f;
  return z;
}

__device__ __forceinline__ v8f wmma_bf16(v16bf a, v16bf b, v8f c) {
  // D = A(16x32 bf16) x B(32x16 bf16) + C(16x16 f32)
  return __builtin_amdgcn_wmma_f32_16x16x32_bf16(false, a, false, b, (short)0, c, false, false);
}

// Load a 16x32 bf16 A/B fragment slice for this lane from a row-major row ptr.
// Lanes 0-15 hold K = {k0+0..7, k0+16..23}; lanes 16-31 hold K = {k0+8..15, k0+24..31}.
__device__ __forceinline__ v16bf bf_frag(const bf16_t* rowptr, int k0, int lane) {
  int koff = (lane & 16) ? 8 : 0;
  union { v16bf v; bf16x8 h[2]; } u;
  u.h[0] = *(const bf16x8*)(rowptr + k0 + koff);
  u.h[1] = *(const bf16x8*)(rowptr + k0 + 16 + koff);
  return u.v;
}

// Same fragment pattern but source is fp32 (converted to bf16 on the fly).
__device__ __forceinline__ v16bf f32_frag(const float* rowptr, int k0, int lane) {
  int koff = (lane & 16) ? 8 : 0;
  const float* a0 = rowptr + k0 + koff;
  const float* a1 = a0 + 16;
  v16bf r;
#pragma unroll
  for (int i = 0; i < 8; ++i) { r[i] = (bf16_t)a0[i]; r[i + 8] = (bf16_t)a1[i]; }
  return r;
}

// Fragment for a K=16 operand zero-padded to K=32 (K>=16 slots are zero).
__device__ __forceinline__ v16bf fragK16(const bf16_t* rowptr, int lane) {
  int koff = (lane & 16) ? 8 : 0;
  union { v16bf v; bf16x8 h[2]; } u;
  u.h[0] = *(const bf16x8*)(rowptr + koff);
#pragma unroll
  for (int i = 0; i < 8; ++i) u.v[8 + i] = (bf16_t)0.f;
  return u.v;
}

// ---------------- prep kernels (weights -> bf16 workspace) ----------------

__global__ void prep_weights(const float* __restrict__ gates,
                             const float* __restrict__ W1,
                             const float* __restrict__ W2,
                             const float* __restrict__ C,
                             const float* __restrict__ Dmat,
                             bf16_t* __restrict__ sW1, bf16_t* __restrict__ sW2,
                             bf16_t* __restrict__ sC,  bf16_t* __restrict__ sD) {
  int i = blockIdx.x * blockDim.x + threadIdx.x;
  const int n1 = N_BR * MH * MEAS_IN;   // 163840: gate-folded W1
  const int n2 = N_BR * RH * MH;        //  32768: W2
  const int n3 = 32 * Dd;               //   8192: C padded to 32 rows
  const int n4 = 32 * UD;               //    512: D padded to 32 rows
  if (i < n1) {
    int k = i % MEAS_IN;
    int n = i / (MH * MEAS_IN);
    float s = (k < Dd) ? (1.f / (1.f + expf(-gates[n * Dd + k]))) : 1.f;
    sW1[i] = (bf16_t)(W1[i] * s);
  } else if (i < n1 + n2) {
    int j = i - n1;
    sW2[j] = (bf16_t)W2[j];
  } else if (i < n1 + n2 + n3) {
    int j = i - n1 - n2;
    int r = j / Dd, c = j % Dd;
    sC[j] = (r < NOBS) ? (bf16_t)C[r * Dd + c] : (bf16_t)0.f;
  } else if (i < n1 + n2 + n3 + n4) {
    int j = i - n1 - n2 - n3;
    int r = j / UD, u = j % UD;
    sD[j] = (r < NOBS) ? (bf16_t)Dmat[r * UD + u] : (bf16_t)0.f;
  }
}

// Fold Bmat into Wout:  M_n = Wout_n @ Bmat_n[:, :RH]   [256 x 64]
//                       Mu  = sum_n Wout_n @ Bmat_n[:, RH:]  [256 x 16]
__global__ void prep_fold(const float* __restrict__ Bmat,
                          const float* __restrict__ Wout,
                          bf16_t* __restrict__ sM, bf16_t* __restrict__ sMu) {
  int i = blockIdx.x * blockDim.x + threadIdx.x;
  const int nM = N_BR * Dd * RH;  // 65536
  if (i < nM) {
    int r = i % RH;
    int d = (i / RH) % Dd;
    int n = i / (RH * Dd);
    float acc = 0.f;
    for (int h2 = 0; h2 < RH; ++h2)
      acc += Wout[((size_t)n * Dd + d) * RH + h2] *
             Bmat[((size_t)n * RH + h2) * (RH + UD) + r];
    sM[i] = (bf16_t)acc;
  } else if (i < nM + Dd * UD) {
    int j = i - nM;
    int u = j % UD, d = j / UD;
    float acc = 0.f;
    for (int n = 0; n < N_BR; ++n)
      for (int h2 = 0; h2 < RH; ++h2)
        acc += Wout[((size_t)n * Dd + d) * RH + h2] *
               Bmat[((size_t)n * RH + h2) * (RH + UD) + RH + u];
    sMu[j] = (bf16_t)acc;
  }
}

// ---------------- fused main kernel ----------------
// One wave32 per 16-row batch tile; 4 waves per block.

__global__ __launch_bounds__(128) void skolr_fused(
    const float* __restrict__ z_dyn, const float* __restrict__ z_static,
    const float* __restrict__ dt, const float* __restrict__ ut,
    const float* __restrict__ b1, const float* __restrict__ gamma_,
    const float* __restrict__ beta_, const float* __restrict__ b2,
    const bf16_t* __restrict__ sW1, const bf16_t* __restrict__ sW2,
    const bf16_t* __restrict__ sM, const bf16_t* __restrict__ sMu,
    const bf16_t* __restrict__ sC, const bf16_t* __restrict__ sD,
    float* __restrict__ out) {
  __shared__ __align__(16) bf16_t smem[4][WAVE_LDS];
  const int lane = threadIdx.x & 31;
  const int wv   = threadIdx.x >> 5;
  const int rA   = lane & 15;   // fragment row / output column within tile
  const int grp  = lane >> 4;   // half-wave group (C/D rows m = v + 8*grp)
  const long rowBase = (long)(blockIdx.x * 4 + wv) * 16;
  const long gr = rowBase + rA; // global batch row this lane serves for A-frags
  bf16_t* st = smem[wv];
  const float dtv = dt[0];

  // Persistent A-fragments of x = [z_dyn | z_static], shared by all branches
  // (spectral gates are folded into sW1 columns).
  v16bf xa[10];
#pragma unroll
  for (int c = 0; c < 8; ++c) xa[c] = f32_frag(z_dyn + gr * Dd, c * 32, lane);
  xa[8] = f32_frag(z_static + gr * SDp, 0, lane);
  xa[9] = f32_frag(z_static + gr * SDp, 32, lane);

  // z_next accumulators: 16 column tiles of 16, f32 (accumulate across branches)
  v8f zacc[16];
#pragma unroll
  for (int t = 0; t < 16; ++t) zacc[t] = zero8();

#pragma unroll 1
  for (int n = 0; n < N_BR; ++n) {
    // ---- GEMM1: h[16,128] = x @ W1'^T + b1 ----
    v8f h[8];
#pragma unroll
    for (int t = 0; t < 8; ++t) {
      v8f a = zero8();
#pragma unroll
      for (int c = 0; c < 10; ++c) {
        v16bf bw = bf_frag(sW1 + (size_t)(n * MH + t * 16 + rA) * MEAS_IN, c * 32, lane);
        a = wmma_bf16(xa[c], bw, a);
      }
      float bb = b1[n * MH + t * 16 + rA];
#pragma unroll
      for (int v = 0; v < 8; ++v) a[v] += bb;
      h[t] = a;
    }

    // ---- LayerNorm over the 128 columns (per row m = v + 8*grp) ----
    float mean[8], rstd[8];
#pragma unroll
    for (int v = 0; v < 8; ++v) {
      float p = 0.f, q = 0.f;
#pragma unroll
      for (int t = 0; t < 8; ++t) { float x = h[t][v]; p += x; q += x * x; }
#pragma unroll
      for (int m = 1; m < 16; m <<= 1) {  // reduce within 16-lane half-group
        p += __shfl_xor(p, m, 32);
        q += __shfl_xor(q, m, 32);
      }
      float mu  = p * (1.f / 128.f);
      float var = q * (1.f / 128.f) - mu * mu;
      mean[v] = mu;
      rstd[v] = rsqrtf(var + 1e-5f);
    }

    // ---- scale/shift + exact GELU, stage h (bf16) to LDS ----
#pragma unroll
    for (int t = 0; t < 8; ++t) {
      float ga = gamma_[n * MH + t * 16 + rA];
      float be = beta_[n * MH + t * 16 + rA];
#pragma unroll
      for (int v = 0; v < 8; ++v) {
        float x = (h[t][v] - mean[v]) * rstd[v] * ga + be;
        x = 0.5f * x * (1.f + erff(x * 0.70710678118654752f));
        st[(v + 8 * grp) * H_STRIDE + t * 16 + rA] = (bf16_t)x;
      }
    }
    __builtin_amdgcn_wave_barrier();

    // ---- GEMM2: g[16,64] = gelu(h) @ W2^T + b2 ----
    v16bf haf[4];
#pragma unroll
    for (int c = 0; c < 4; ++c) haf[c] = bf_frag(st + rA * H_STRIDE, c * 32, lane);
    v8f g4[4];
#pragma unroll
    for (int t = 0; t < 4; ++t) {
      v8f a = zero8();
#pragma unroll
      for (int c = 0; c < 4; ++c) {
        v16bf bw = bf_frag(sW2 + (size_t)(n * RH + t * 16 + rA) * MH, c * 32, lane);
        a = wmma_bf16(haf[c], bw, a);
      }
      float bb = b2[n * RH + t * 16 + rA];
#pragma unroll
      for (int v = 0; v < 8; ++v) a[v] += bb;
      g4[t] = a;
    }
    __builtin_amdgcn_wave_barrier();

    // ---- stage g (bf16) ----
#pragma unroll
    for (int t = 0; t < 4; ++t)
#pragma unroll
      for (int v = 0; v < 8; ++v)
        st[(v + 8 * grp) * G_STRIDE + t * 16 + rA] = (bf16_t)g4[t][v];
    __builtin_amdgcn_wave_barrier();

    // ---- GEMM3: zacc += g @ M_n^T  (M_n = Wout_n @ BmatG_n, pre-folded) ----
    v16bf gaf0 = bf_frag(st + rA * G_STRIDE, 0, lane);
    v16bf gaf1 = bf_frag(st + rA * G_STRIDE, 32, lane);
#pragma unroll
    for (int t = 0; t < 16; ++t) {
      const bf16_t* mrow = sM + (size_t)(n * Dd + t * 16 + rA) * RH;
      zacc[t] = wmma_bf16(gaf0, bf_frag(mrow, 0, lane), zacc[t]);
      zacc[t] = wmma_bf16(gaf1, bf_frag(mrow, 32, lane), zacc[t]);
    }
    __builtin_amdgcn_wave_barrier();
  }

  // ---- ut*dt contribution: zacc += ut_dt @ Mu^T ----
  v16bf uf;
  {
    const float* up = ut + gr * UD;
    int koff = (lane & 16) ? 8 : 0;
#pragma unroll
    for (int i = 0; i < 8; ++i) {
      uf[i]     = (bf16_t)(up[koff + i] * dtv);
      uf[i + 8] = (bf16_t)0.f;  // K=16 zero-padded to 32
    }
  }
#pragma unroll
  for (int t = 0; t < 16; ++t)
    zacc[t] = wmma_bf16(uf, fragK16(sMu + (size_t)(t * 16 + rA) * UD, lane), zacc[t]);

  // ---- store z_next (fp32) and stage z (bf16) for the yt GEMM ----
  float* out_z = out;
  float* out_y = out + (size_t)BSZ * Dd;
#pragma unroll
  for (int t = 0; t < 16; ++t)
#pragma unroll
    for (int v = 0; v < 8; ++v) {
      int m = v + 8 * grp;
      float val = zacc[t][v];
      out_z[(rowBase + m) * Dd + t * 16 + rA] = val;
      st[m * Z_STRIDE + t * 16 + rA] = (bf16_t)val;
    }
  __builtin_amdgcn_wave_barrier();

  // ---- GEMM4: yt[16,25] = z_next @ C^T + ut_dt @ D^T (N padded to 32) ----
  v8f y0 = zero8(), y1 = zero8();
#pragma unroll
  for (int c = 0; c < 8; ++c) {
    v16bf zf = bf_frag(st + rA * Z_STRIDE, c * 32, lane);
    y0 = wmma_bf16(zf, bf_frag(sC + (size_t)(rA) * Dd,      c * 32, lane), y0);
    y1 = wmma_bf16(zf, bf_frag(sC + (size_t)(16 + rA) * Dd, c * 32, lane), y1);
  }
  y0 = wmma_bf16(uf, fragK16(sD + (size_t)rA * UD,        lane), y0);
  y1 = wmma_bf16(uf, fragK16(sD + (size_t)(16 + rA) * UD, lane), y1);
#pragma unroll
  for (int v = 0; v < 8; ++v) {
    int m = v + 8 * grp;
    out_y[(rowBase + m) * NOBS + rA] = y0[v];            // cols 0..15 always valid
    int c1 = 16 + rA;
    if (c1 < NOBS) out_y[(rowBase + m) * NOBS + c1] = y1[v];
  }
}

// ---------------- launch ----------------

extern "C" void kernel_launch(void* const* d_in, const int* in_sizes, int n_in,
                              void* d_out, int out_size, void* d_ws, size_t ws_size,
                              hipStream_t stream) {
  const float* z_dyn    = (const float*)d_in[0];
  const float* z_static = (const float*)d_in[1];
  const float* dt       = (const float*)d_in[2];
  const float* ut       = (const float*)d_in[3];
  const float* gates    = (const float*)d_in[4];
  const float* W1       = (const float*)d_in[5];
  const float* b1       = (const float*)d_in[6];
  const float* gamma_   = (const float*)d_in[7];
  const float* beta_    = (const float*)d_in[8];
  const float* W2       = (const float*)d_in[9];
  const float* b2       = (const float*)d_in[10];
  // d_in[11] lam_real, d_in[12] lam_imag: unused — h0==0 makes them vanish.
  const float* Bmat     = (const float*)d_in[13];
  const float* Wout     = (const float*)d_in[14];
  const float* C        = (const float*)d_in[15];
  const float* Dmat     = (const float*)d_in[16];

  char* ws = (char*)d_ws;
  bf16_t* sW1 = (bf16_t*)(ws);            // 4*128*320*2 = 327680 B
  bf16_t* sW2 = (bf16_t*)(ws + 327680);   // 4*64*128*2  =  65536 B
  bf16_t* sM  = (bf16_t*)(ws + 393216);   // 4*256*64*2  = 131072 B
  bf16_t* sMu = (bf16_t*)(ws + 524288);   // 256*16*2    =   8192 B
  bf16_t* sC  = (bf16_t*)(ws + 532480);   // 32*256*2    =  16384 B
  bf16_t* sD  = (bf16_t*)(ws + 548864);   // 32*16*2     =   1024 B

  float* out = (float*)d_out;

  const int nconv = 4 * 128 * 320 + 4 * 64 * 128 + 32 * 256 + 32 * 16;  // 205312
  prep_weights<<<(nconv + 255) / 256, 256, 0, stream>>>(gates, W1, W2, C, Dmat,
                                                        sW1, sW2, sC, sD);
  const int nfold = 4 * 256 * 64 + 256 * 16;  // 69632
  prep_fold<<<(nfold + 255) / 256, 256, 0, stream>>>(Bmat, Wout, sM, sMu);

  // one wave per 16 batch rows, 4 waves per block -> 64 rows/block
  skolr_fused<<<BSZ / 64, 128, 0, stream>>>(z_dyn, z_static, dt, ut, b1, gamma_,
                                            beta_, b2, sW1, sW2, sM, sMu, sC, sD,
                                            out);
}
